// WindowAttention_6597069767438
// MI455X (gfx1250) — matrix operand order repaired and checked by reading, not compiled, not run on hardware
//
#include <hip/hip_runtime.h>
#include <hip/hip_bf16.h>

typedef __attribute__((ext_vector_type(16))) _Float16 v16h;
typedef __attribute__((ext_vector_type(8)))  _Float16 v8h;
typedef __attribute__((ext_vector_type(4)))  _Float16 v4h;
typedef __attribute__((ext_vector_type(8)))  float    v8f;
typedef __attribute__((ext_vector_type(4)))  float    v4f;

#define NUM_HEADS 4
#define HEAD_DIM  32
#define SEQ       49
#define CDIM      128
#define SCALE     0.17677669529663687f   // 32^-0.5

#define LDX 136   // row stride (halfs) for 64x128 tiles, 16B multiple, odd multiple of 16B (17)
#define LDV 72    // row stride for V^T (128 x 64)
#define LDP 72    // row stride for prob tiles (64 x 64)

union F16x16 { v16h v; v8h h[2]; };

// A-matrix 16x32 f16 fragment from row-major storage (ISA 7.12.2):
// lane r (0-15): M=r, K=+0..7 then +16..23 ; lane r+16: K=+8..15 then +24..31
__device__ __forceinline__ v16h load_a16(const _Float16* base, int ld, int lane) {
  const _Float16* p = base + (lane & 15) * ld + ((lane >> 4) & 1) * 8;
  F16x16 f;
  f.h[0] = *(const v8h*)(p);
  f.h[1] = *(const v8h*)(p + 16);
  return f.v;
}

// B-matrix 32x16 f16 fragment from TRANSPOSED (NxK row-major) storage:
// lane l (0-15): N=l, K=0..7 then 16..23 ; lane l+16: N=l, K=8..15 then 24..31
__device__ __forceinline__ v16h load_b16(const _Float16* baseT, int ld, int lane) {
  const _Float16* p = baseT + (lane & 15) * ld + ((lane >> 4) & 1) * 8;
  F16x16 f;
  f.h[0] = *(const v8h*)(p);
  f.h[1] = *(const v8h*)(p + 16);
  return f.v;
}

__device__ __forceinline__ v8f wmma16(v16h a, v16h b, v8f c) {
  return __builtin_amdgcn_wmma_f32_16x16x32_f16(false, a, false, b, (short)0, c, false, false);
}

__device__ __forceinline__ v8h pack8(v8f a) {
  v8h h;
  #pragma unroll
  for (int j = 0; j < 8; ++j) h[j] = (_Float16)a[j];
  return h;
}

// ---- prep: weights -> f16 [N][K] (Q rows pre-scaled); mask -> padded+permuted;
//      qkv bias -> pre-scaled f32 copy ----
__global__ void prep_kernel(const float* __restrict__ qkv_w,
                            const float* __restrict__ proj_w,
                            const float* __restrict__ attn_mask,
                            const float* __restrict__ qkv_b,
                            _Float16* __restrict__ wqkvT,
                            _Float16* __restrict__ wprojT,
                            float* __restrict__ mpad,
                            float* __restrict__ qkv_bs) {
  int i = blockIdx.x * blockDim.x + threadIdx.x;
  if (i < 384 * 128) {
    int n = i >> 7, k = i & 127;
    float s = (n < 128) ? SCALE : 1.0f;          // fold attention scale into Wq
    wqkvT[n * 128 + k] = (_Float16)(qkv_w[k * 384 + n] * s);
  }
  if (i < 128 * 128) {
    int n = i >> 7, k = i & 127;
    wprojT[n * 128 + k] = (_Float16)proj_w[k * 128 + n];
  }
  if (i < 384) qkv_bs[i] = qkv_b[i] * ((i < 128) ? SCALE : 1.0f);
  if (i < 64 * 64 * 64) {
    int w = i >> 12, r = (i >> 6) & 63, c = i & 63;
    float v;
    if (c < SEQ) v = (r < SEQ) ? attn_mask[(w * SEQ + r) * SEQ + c] : 0.0f;
    else         v = -1e30f;                     // padded key columns -> exp() == 0
    // permuted layout [w][row][colw][nT] so each lane reads one v4f per row
    mpad[(w << 12) + (r << 6) + ((c & 15) << 2) + (c >> 4)] = v;
  }
}

// ---- fused window attention: one workgroup (8 wave32) per window ----
__global__ __launch_bounds__(256) void window_attn_kernel(
    const float* __restrict__ x, const float* __restrict__ mpad,
    const float* __restrict__ qkv_bs, const float* __restrict__ proj_b,
    const _Float16* __restrict__ wqkvT, const _Float16* __restrict__ wprojT,
    float* __restrict__ out)
{
  __shared__ _Float16 s_xo[64 * LDX];            // x (stage A/B), then attn-out (stage C/D)
  __shared__ _Float16 s_q [64 * LDX];            // Q * scale, [token][h*32+d]
  __shared__ _Float16 s_k [64 * LDX];            // K,          [token][h*32+d]
  __shared__ _Float16 s_vt[128 * LDV];           // V^T,        [h*32+d][token]
  __shared__ _Float16 s_p [NUM_HEADS * 64 * LDP];// probs,      [h][token][key]

  const int tid  = threadIdx.x;
  const int lane = tid & 31;
  const int wave = tid >> 5;
  const int colw = lane & 15;
  const int hsel = (lane >> 4) & 1;
  const int bIdx = blockIdx.x;
  const int wIdx = bIdx & 63;                    // window id = b % nW

  // ---- Stage A: x[b] -> LDS f16 (float4 loads), zero-pad rows 49..63 ----
  const float* xb = x + (size_t)bIdx * SEQ * CDIM;
  for (int i = tid; i < 64 * 32; i += 256) {     // 64 rows x 32 float4
    int r = i >> 5, c4 = i & 31;
    v4f v = {0.0f, 0.0f, 0.0f, 0.0f};
    if (r < SEQ) v = ((const v4f*)xb)[r * 32 + c4];
    v4h h;
    #pragma unroll
    for (int j = 0; j < 4; ++j) h[j] = (_Float16)v[j];
    *(v4h*)(s_xo + r * LDX + c4 * 4) = h;
  }
  __syncthreads();

  // ---- Stage B1 (flipped): [q|k]^T tiles = Wqkv^T @ x^T ; contiguous b128 stores ----
  for (int t = wave; t < 64; t += 8) {           // 16 feature-tiles x 4 token-tiles
    int m0 = (t >> 2) * 16;                      // feature tile base (0..240)
    int n0 = (t & 3) * 16;                       // token tile base (fixed per wave)
    v8f acc = *(const v8f*)(qkv_bs + m0 + hsel * 8);
    #pragma unroll
    for (int k = 0; k < 4; ++k) {
      v16h a = load_a16(wqkvT + m0 * 128 + k * 32, 128, lane);   // weights (global)
      v16h b = load_b16(s_xo + n0 * LDX + k * 32, LDX, lane);    // x tokens (LDS)
      acc = wmma16(a, b, acc);
    }
    int tok = n0 + colw;
    int fb  = m0 + hsel * 8;                     // 8 consecutive features
    _Float16* dst = (m0 < 128) ? (s_q + tok * LDX + fb)
                               : (s_k + tok * LDX + (fb - 128));
    *(v8h*)dst = pack8(acc);
  }
  // ---- Stage B2 (normal): V tiles = x @ Wv ; g-dim walks tokens -> contiguous in V^T ----
  for (int t = wave; t < 32; t += 8) {           // 4 token-tiles x 8 feature-tiles
    int m0 = (t & 3) * 16;                       // token tile base (fixed per wave)
    int n0 = 256 + (t >> 2) * 16;                // V feature tile base
    float bias = qkv_bs[n0 + colw];
    v8f acc;
    #pragma unroll
    for (int g = 0; g < 8; ++g) acc[g] = bias;
    #pragma unroll
    for (int k = 0; k < 4; ++k) {
      v16h a = load_a16(s_xo + m0 * LDX + k * 32, LDX, lane);    // x tokens (LDS)
      v16h b = load_b16(wqkvT + n0 * 128 + k * 32, 128, lane);   // Wv^T (global)
      acc = wmma16(a, b, acc);
    }
    // C rows (g) = 8 consecutive tokens of d-row (n0-256+colw)
    *(v8h*)(s_vt + (n0 - 256 + colw) * LDV + m0 + hsel * 8) = pack8(acc);
  }
  __syncthreads();

  // ---- Stage C: per (head, 16-token strip): scores -> softmax (in regs) -> V^T@P^T ----
  for (int p = wave; p < 16; p += 8) {
    int head = p >> 2, m0 = (p & 3) * 16;        // m0 = token strip base
    v16h aq = load_a16(s_q + m0 * LDX + head * HEAD_DIM, LDX, lane);
    v8f ct[4];
    #pragma unroll
    for (int nT = 0; nT < 4; ++nT) {
      v16h bk = load_b16(s_k + nT * 16 * LDX + head * HEAD_DIM, LDX, lane);
      v8f z;
      #pragma unroll
      for (int g = 0; g < 8; ++g) z[g] = 0.0f;
      ct[nT] = wmma16(aq, bk, z);                // q.k^T * scale (scale folded into Wq)
    }
    // add padded window mask: one v4f per row per lane
    const float* mp = mpad + (wIdx << 12);
    #pragma unroll
    for (int g = 0; g < 8; ++g) {
      int row = m0 + g + hsel * 8;
      v4f mrow = *(const v4f*)(mp + (row << 6) + (colw << 2));
      #pragma unroll
      for (int nT = 0; nT < 4; ++nT) ct[nT][g] += mrow[nT];
    }
    // row-wise softmax: row lives in one VGPR across a 16-lane half
    #pragma unroll
    for (int g = 0; g < 8; ++g) {
      float mx = fmaxf(fmaxf(ct[0][g], ct[1][g]), fmaxf(ct[2][g], ct[3][g]));
      mx = fmaxf(mx, __shfl_xor(mx, 1, 32));
      mx = fmaxf(mx, __shfl_xor(mx, 2, 32));
      mx = fmaxf(mx, __shfl_xor(mx, 4, 32));
      mx = fmaxf(mx, __shfl_xor(mx, 8, 32));
      float s = 0.0f;
      #pragma unroll
      for (int nT = 0; nT < 4; ++nT) {
        float e = __expf(ct[nT][g] - mx);
        ct[nT][g] = e; s += e;
      }
      s += __shfl_xor(s, 1, 32);
      s += __shfl_xor(s, 2, 32);
      s += __shfl_xor(s, 4, 32);
      s += __shfl_xor(s, 8, 32);
      float inv = 1.0f / s;
      int row = m0 + g + hsel * 8;
      #pragma unroll
      for (int nT = 0; nT < 4; ++nT)
        s_p[(head * 64 + row) * LDP + nT * 16 + colw] = (_Float16)(ct[nT][g] * inv);
    }
    // out^T = V^T @ P^T : rows = d, cols = this wave's 16 tokens (own prob rows only)
    #pragma unroll
    for (int md = 0; md < 2; ++md) {
      v8f acc;
      #pragma unroll
      for (int g = 0; g < 8; ++g) acc[g] = 0.0f;
      #pragma unroll
      for (int kk = 0; kk < 2; ++kk) {
        v16h av = load_a16(s_vt + (head * HEAD_DIM + md * 16) * LDV + kk * 32, LDV, lane);
        v16h bp = load_b16(s_p + (head * 64 + m0) * LDP + kk * 32, LDP, lane);
        acc = wmma16(av, bp, acc);
      }
      int tok = m0 + colw;
      int c0  = head * HEAD_DIM + md * 16 + hsel * 8;   // 8 consecutive channels
      *(v8h*)(s_xo + tok * LDX + c0) = pack8(acc);      // attn-out, [token][c]
    }
  }
  __syncthreads();

  // ---- Stage D (flipped): out^T = Wproj^T @ attnout^T ; rows = channels, cols = tokens ----
  float* ob = out + (size_t)bIdx * SEQ * CDIM;
  for (int t = wave; t < 32; t += 8) {           // 8 channel-tiles x 4 token-tiles
    int m0 = (t >> 2) * 16;                      // channel tile base
    int n0 = (t & 3) * 16;                       // token tile base
    v8f acc = *(const v8f*)(proj_b + m0 + hsel * 8);
    #pragma unroll
    for (int k = 0; k < 4; ++k) {
      v16h a = load_a16(wprojT + m0 * 128 + k * 32, 128, lane);
      v16h b = load_b16(s_xo + n0 * LDX + k * 32, LDX, lane);
      acc = wmma16(a, b, acc);
    }
    int tok = n0 + colw;
    if (tok < SEQ)                               // 8 consecutive channels of one token
      *(v8f*)(ob + tok * CDIM + m0 + hsel * 8) = acc;
  }
}

extern "C" void kernel_launch(void* const* d_in, const int* in_sizes, int n_in,
                              void* d_out, int out_size, void* d_ws, size_t ws_size,
                              hipStream_t stream) {
  const float* x         = (const float*)d_in[0];
  const float* attn_mask = (const float*)d_in[1];
  const float* qkv_w     = (const float*)d_in[2];
  const float* qkv_b     = (const float*)d_in[3];
  const float* proj_w    = (const float*)d_in[4];
  const float* proj_b    = (const float*)d_in[5];
  float* out = (float*)d_out;

  _Float16* wqkvT  = (_Float16*)d_ws;                    // [384][128] f16 (Wq pre-scaled)
  _Float16* wprojT = wqkvT + 384 * 128;                  // [128][128] f16
  float*    mpad   = (float*)(wprojT + 128 * 128);       // [64][64][16][4] f32 padded mask
  float*    qkv_bs = mpad + 64 * 64 * 64;                // [384] f32 (Q part pre-scaled)

  prep_kernel<<<1024, 256, 0, stream>>>(qkv_w, proj_w, attn_mask, qkv_b,
                                        wqkvT, wprojT, mpad, qkv_bs);
  window_attn_kernel<<<2048, 256, 0, stream>>>(x, mpad, qkv_bs, proj_b,
                                               wqkvT, wprojT, out);
}
